// ModulatedPEConv2d_38130719654082
// MI455X (gfx1250) — matrix operand order repaired and checked by reading, not compiled, not run on hardware
//
#include <hip/hip_runtime.h>
#include <hip/hip_bf16.h>
#include <stdint.h>

// ---------- CDNA5 fragment types (wave32 WMMA) ----------
typedef __bf16 v16bf __attribute__((ext_vector_type(16)));
typedef float  v8f   __attribute__((ext_vector_type(8)));

#define N_B    8
#define CIN    512
#define COUT   512
#define H_IN   32
#define STY    512
#define OHW    65     // transposed-conv output 65x65
#define OUT_HW 64     // post-blur output 64x64

#define STY_SCALE 0.04419417382415922f   // 1/sqrt(512)
#define W_SCALE   0.014731391274719742f  // 1/sqrt(512*9)
#define WSC2      2.170138888888889e-4f  // 1/4608

__device__ __forceinline__ __bf16 f2bf(float f) {
  unsigned u = __builtin_bit_cast(unsigned, f);
  u += 0x7FFFu + ((u >> 16) & 1u);                 // round-to-nearest-even
  unsigned short h = (unsigned short)(u >> 16);
  return __builtin_bit_cast(__bf16, h);
}

__device__ __forceinline__ float wave_sum(float v) {
#pragma unroll
  for (int off = 16; off > 0; off >>= 1) v += __shfl_xor(v, off, 32);
  return v;
}

// ---------- Stage 1: s[n][cin] = style . style_w^T * sty_scale + style_b ----------
__global__ __launch_bounds__(128) void style_kernel(
    const float* __restrict__ style, const float* __restrict__ style_w,
    const float* __restrict__ style_b, float* __restrict__ s) {
  int wid  = (blockIdx.x * 128 + threadIdx.x) >> 5;   // 4096 waves
  int lane = threadIdx.x & 31;
  int n = wid >> 9, ci = wid & 511;
  float sum = 0.f;
  for (int j = lane; j < STY; j += 32)
    sum += style[n * STY + j] * style_w[ci * STY + j];
  sum = wave_sum(sum);
  if (lane == 0) s[wid] = sum * STY_SCALE + style_b[ci];
}

// ---------- Stage 2a: wsq[co][cin] = sum_{3x3} weight^2 ----------
__global__ __launch_bounds__(256) void wsq_kernel(
    const float* __restrict__ weight, float* __restrict__ wsq) {
  int tid = blockIdx.x * 256 + threadIdx.x;           // 262144
  if (tid >= COUT * CIN) return;
  const float* wp = weight + (size_t)tid * 9;
  float acc = 0.f;
#pragma unroll
  for (int k = 0; k < 9; k++) acc += wp[k] * wp[k];
  wsq[tid] = acc;
}

// ---------- Stage 2b: demod[n][co] = rsqrt(wscale^2 * sum_cin s^2*wsq + eps) ----------
__global__ __launch_bounds__(128) void demod_kernel(
    const float* __restrict__ s, const float* __restrict__ wsq,
    float* __restrict__ demod) {
  int wid  = (blockIdx.x * 128 + threadIdx.x) >> 5;   // 4096 waves = (n,co)
  int lane = threadIdx.x & 31;
  int n = wid >> 9, co = wid & 511;
  float sum = 0.f;
  for (int ci = lane; ci < CIN; ci += 32) {
    float sv = s[n * CIN + ci];
    sum += sv * sv * wsq[co * CIN + ci];
  }
  sum = wave_sum(sum);
  if (lane == 0) demod[wid] = rsqrtf(sum * WSC2 + 1e-8f);
}

// ---------- Stage 3: xT[n][h][w][cin] = bf16(x[n][cin][h][w]) (LDS transpose) ----------
__global__ __launch_bounds__(1024) void xt_kernel(
    const float* __restrict__ x, __bf16* __restrict__ xT) {
  __shared__ float tile[32][33];
  int n = blockIdx.z, i = blockIdx.y, cb = blockIdx.x * 32;
  int tx = threadIdx.x, ty = threadIdx.y;
  tile[ty][tx] = x[((size_t)(n * CIN + cb + ty) * H_IN + i) * H_IN + tx];
  __syncthreads();
  xT[((size_t)((n * H_IN + i) * H_IN) + ty) * CIN + cb + tx] = f2bf(tile[tx][ty]);
}

// ---------- Stage 3b: zero the B guard pixel (512 bf16 = 256 dwords) ----------
__global__ void zpad_kernel(unsigned* __restrict__ xzero) {
  xzero[threadIdx.x] = 0u;
}

// ---------- Stage 4: fragment-ordered modulated weights ----------
// wb2[n][tap][co16][cin32][lane][16] : lane's exact A-fragment payload for
// V_WMMA_F32_16X16X32_BF16 (M = lane&15, K runs [khalf*16,+16), the k order this kernel uses for A and B).
__global__ __launch_bounds__(256) void modw_kernel(
    const float* __restrict__ weight, const float* __restrict__ s,
    const float* __restrict__ demod, __bf16* __restrict__ wb2) {
  unsigned tid = blockIdx.x * 256 + threadIdx.x;      // 18,874,368
  if (tid >= (unsigned)N_B * 9 * COUT * CIN) return;
  int v    = tid & 15;          // element within lane fragment
  int lane = (tid >> 4) & 31;
  int cbi  = (tid >> 9) & 15;   // cin block (32)
  int cob  = (tid >> 13) & 31;  // cout block (16)
  unsigned t5 = tid >> 18;      // n*9 + tap, [0,72)
  int tap = t5 % 9;
  int n   = t5 / 9;
  int kh = tap / 3, kw = tap % 3;
  int m = lane & 15, khalf = lane >> 4;
  int kk  = khalf * 16 + v;
  int cin = cbi * 32 + kk;
  int co  = cob * 16 + m;
  float val = weight[((size_t)(co * CIN + cin) * 3 + (2 - kh)) * 3 + (2 - kw)];
  val *= W_SCALE * s[n * CIN + cin] * demod[n * COUT + co];
  wb2[tid] = f2bf(val);
}

// ---------- Stage 5: parity-decomposed implicit GEMM transposed conv (WMMA) ----------
// grid: x = 65 rows * 5 col-tiles, y = COUT/128, z = n. block = 256 (8 waves).
__global__ __launch_bounds__(256) void conv_kernel(
    const __bf16* __restrict__ xT, const __bf16* __restrict__ wb2,
    float* __restrict__ y) {
  const int lane  = threadIdx.x & 31;
  const int wave  = threadIdx.x >> 5;
  const int m     = lane & 15;        // A: M row, B: N col, store: position
  const int khalf = lane >> 4;        // K-half selector for B fragment
  const int n     = blockIdx.z;
  const int cob   = blockIdx.y * 8 + wave;   // 16-wide cout block id
  const __bf16* xzero = xT + (size_t)N_B * H_IN * H_IN * CIN;  // 512 zeros

  int t  = blockIdx.x;
  int oh = t / 5, r5 = t % 5;
  int pw    = (r5 < 3) ? 0 : 1;                 // output-column parity
  int c0    = (pw ? (r5 - 3) : r5) * 16;        // tile start (parity-class idx)
  int ncols = pw ? 32 : 33;
  int c     = c0 + m;                            // this lane's class-column
  int ph    = oh & 1;                            // output-row parity

  v8f acc = {};
  const int khs = ph ? 1 : 0, khe = ph ? 1 : 2;
  const int kws = pw ? 1 : 0, kwe = pw ? 1 : 2;

  for (int kh = khs; kh <= khe; kh += 2) {
    int i = (oh + kh - 2) >> 1;                  // dilated-input row (uniform)
    if (i < 0 || i > 31) continue;
    for (int kw = kws; kw <= kwe; kw += 2) {
      int j = ((2 * c + pw) + kw - 2) >> 1;      // per-lane input column
      bool jv = (j >= 0) && (j < 32);
      // A: fragment-ordered weights, one contiguous 32B per lane, lane-major
      const __bf16* wp = wb2
          + ((size_t)((n * 9 + kh * 3 + kw) * 32 + cob) * 16) * 512
          + lane * 16;
      // B: 16 consecutive cin at this lane's pixel; invalid -> zero guard pixel
      const __bf16* xp = jv
          ? (xT + ((size_t)((n * H_IN + i) * H_IN + j)) * CIN + khalf * 16)
          : (xzero + khalf * 16);
#pragma unroll 8
      for (int cbi = 0; cbi < 16; ++cbi) {
        __builtin_prefetch(wp + (size_t)cbi * 512 + 4096, 0, 3);
        v16bf a = *(const v16bf*)(wp + (size_t)cbi * 512);
        v16bf b = *(const v16bf*)(xp + cbi * 32);
        acc = __builtin_amdgcn_wmma_f32_16x16x32_bf16(
                  false, a, false, b, (short)0, acc, false, false);
      }
    }
  }

  // C/D layout: VGPR r -> M = r + khalf*8, N = lane&15
  if (c < ncols) {
    int ow = 2 * c + pw;
    size_t base = ((size_t)(n * COUT + cob * 16 + khalf * 8) * OHW + oh) * OHW + ow;
#pragma unroll
    for (int r = 0; r < 8; r++)
      y[base + (size_t)r * (OHW * OHW)] = acc[r];
  }
}

// ---------- Stage 6: 4x4 blur, pad 1, 65x65 -> 64x64 ----------
__global__ __launch_bounds__(256) void blur_kernel(
    const float* __restrict__ y, float* __restrict__ out) {
  int tid = blockIdx.x * 256 + threadIdx.x;     // 16,777,216
  int c  = tid & 63;
  int r  = (tid >> 6) & 63;
  int nc = tid >> 12;                            // n*COUT + co
  const float* yp = y + (size_t)nc * (OHW * OHW);
  const float k1[4] = {1.f, 3.f, 3.f, 1.f};
  float sum = 0.f;
#pragma unroll
  for (int a = 0; a < 4; a++) {
    int rr = r - 1 + a;
    if (rr < 0 || rr >= OHW) continue;
#pragma unroll
    for (int b = 0; b < 4; b++) {
      int cc = c - 1 + b;
      if (cc < 0 || cc >= OHW) continue;
      sum += yp[rr * OHW + cc] * (k1[a] * k1[b]);
    }
  }
  out[tid] = sum * (1.0f / 16.0f);
}

// ---------- launcher ----------
extern "C" void kernel_launch(void* const* d_in, const int* in_sizes, int n_in,
                              void* d_out, int out_size, void* d_ws, size_t ws_size,
                              hipStream_t stream) {
  const float* x       = (const float*)d_in[0];  // (8,512,32,32)
  const float* style   = (const float*)d_in[1];  // (8,512)
  const float* weight  = (const float*)d_in[2];  // (512,512,3,3)
  const float* style_w = (const float*)d_in[3];  // (512,512)
  const float* style_b = (const float*)d_in[4];  // (512,)
  float* out = (float*)d_out;                    // (8,512,64,64)

  char* ws = (char*)d_ws;
  float*  s     = (float*)(ws + 0);                 //    16 KB
  float*  demod = (float*)(ws + (16u << 10));       //    16 KB
  float*  wsq   = (float*)(ws + (32u << 10));       //     1 MB
  __bf16* xT    = (__bf16*)(ws + 1081344);          //   8 MB + 1KB zero pad
  __bf16* wb2   = (__bf16*)(ws + 9470976);          //  36 MB (fragment order)
  float*  y     = (float*)(ws + 47219712);          //  66 MB (8x512x65x65 f32)
  (void)in_sizes; (void)n_in; (void)out_size; (void)ws_size;

  style_kernel<<<dim3(1024), dim3(128), 0, stream>>>(style, style_w, style_b, s);
  wsq_kernel  <<<dim3(1024), dim3(256), 0, stream>>>(weight, wsq);
  demod_kernel<<<dim3(1024), dim3(128), 0, stream>>>(s, wsq, demod);
  xt_kernel   <<<dim3(16, 32, 8), dim3(32, 32), 0, stream>>>(x, xT);
  zpad_kernel <<<dim3(1), dim3(256), 0, stream>>>(
      (unsigned*)(xT + (size_t)N_B * H_IN * H_IN * CIN));
  modw_kernel <<<dim3(73728), dim3(256), 0, stream>>>(weight, s, demod, wb2);
  conv_kernel <<<dim3(325, 4, 8), dim3(256), 0, stream>>>(xT, wb2, y);
  blur_kernel <<<dim3(65536), dim3(256), 0, stream>>>(y, out);
}
